// LocallyConnected2d_41102837022853
// MI455X (gfx1250) — compile-verified, matrix operand order checked by reading, not run on hardware
//
#include <hip/hip_runtime.h>
#include <hip/hip_bf16.h>

// ---- problem constants (from reference) ----
#define HH    64
#define WW    64
#define CIN   64
#define COUTC 64
#define KR    3
#define HOUT  62
#define WOUT  62
#define BATCH 32
#define KDIM  (CIN * KR * KR)   // 576
#define NCHUNK (KDIM / 32)      // 18 bf16-WMMA K-steps

typedef __attribute__((ext_vector_type(16))) __bf16 v16bf;
typedef __attribute__((ext_vector_type(8)))  float  v8f;
typedef __attribute__((ext_vector_type(4)))  float  v4f;

__device__ __forceinline__ unsigned int f2bf(float f) {
    // round-to-nearest-even fp32 -> bf16 (as uint16 in low bits)
    unsigned int u = __float_as_uint(f);
    unsigned int r = u + 0x7FFFu + ((u >> 16) & 1u);
    return r >> 16;
}

__device__ __forceinline__ v4f ld_nt4(const float* p) {
    // weights are single-use: non-temporal to keep L2 for features/output
    return __builtin_nontemporal_load((const v4f*)p);
}

__global__ __launch_bounds__(128, 4)
void lc2d_wmma_kernel(const float* __restrict__ feat,   // [32,64,64,64]
                      const float* __restrict__ wgt,    // [62,62,64,576]
                      const float* __restrict__ bias,   // [62,62,64]
                      float* __restrict__ out) {        // [32,64,62,62]
    const int h    = blockIdx.y;
    const int w    = blockIdx.x;
    const int hw   = h * WOUT + w;
    const int tid  = threadIdx.x;
    const int lane = tid & 31;
    const int wave = tid >> 5;              // 0..3 = cout tile (N-tile)

    // A matrix (im2col, bf16) staged in LDS in exact WMMA A-fragment order:
    // element slot = ((t*2 + tile_m)*32 + lane)*16 + e   (e = 0..15 bf16/lane)
    __shared__ __align__(16) unsigned int ldsA[BATCH * KDIM / 2];  // 36 KB

    // ---------------- stage 1: cooperative im2col + fp32->bf16 ----------------
    for (int p = tid; p < BATCH * KDIM / 2; p += 128) {
        const int v      = p & 7;           // VGPR index within fragment (pair of bf16)
        const int l      = (p >> 3) & 31;   // destination lane slot
        const int tmc    = p >> 8;          // (t*2 + tile_m)
        const int tile_m = tmc & 1;
        const int t      = tmc >> 1;
        const int half   = l >> 4;
        const int m      = l & 15;
        // intra-lane K layout of 16-bit A fragments (ISA 7.12.2)
        const int k0 = t * 32 + (v < 4 ? 2 * v : 16 + 2 * (v - 4)) + (half ? 8 : 0);
        const int b  = tile_m * 16 + m;
        const int c0 = k0 / 9,       r0 = k0 - c0 * 9;
        const int c1 = (k0 + 1) / 9, r1 = (k0 + 1) - c1 * 9;
        const float f0 = feat[((b * CIN + c0) * HH + (h + r0 / 3)) * WW + (w + r0 % 3)];
        const float f1 = feat[((b * CIN + c1) * HH + (h + r1 / 3)) * WW + (w + r1 % 3)];
        ldsA[p] = f2bf(f0) | (f2bf(f1) << 16);
    }
    __syncthreads();

    // ---------------- stage 2: WMMA GEMM, weights streamed NT ----------------
    const int ncol = lane & 15;
    const int half = lane >> 4;
    const int o    = wave * 16 + ncol;      // this lane's cout column
    const float* wcol = wgt + (size_t)hw * (COUTC * KDIM) + (size_t)o * KDIM + half * 8;

    v8f acc0 = {};   // batch tile 0 (b = 0..15)
    v8f acc1 = {};   // batch tile 1 (b = 16..31)

    for (int t = 0; t < NCHUNK; ++t) {
        // B fragment: 16 contiguous-run fp32 from this lane's weight column
        const float* p0 = wcol + t * 32;
        const v4f b0 = ld_nt4(p0);          // K_local = 8*half + 0..3
        const v4f b1 = ld_nt4(p0 + 4);      // K_local = 8*half + 4..7
        const v4f b2 = ld_nt4(p0 + 16);     // K_local = 16 + 8*half + 0..3
        const v4f b3 = ld_nt4(p0 + 20);     // K_local = 16 + 8*half + 4..7
        v16bf bfrag;
#pragma unroll
        for (int i = 0; i < 4; ++i) {
            bfrag[i]      = (__bf16)b0[i];
            bfrag[4 + i]  = (__bf16)b1[i];
            bfrag[8 + i]  = (__bf16)b2[i];
            bfrag[12 + i] = (__bf16)b3[i];
        }
        // A fragments: aligned 32B reads from LDS (fragment-order layout)
        const v16bf a0 = *(const v16bf*)&ldsA[((t * 2 + 0) * 32 + lane) * 8];
        const v16bf a1 = *(const v16bf*)&ldsA[((t * 2 + 1) * 32 + lane) * 8];

        acc0 = __builtin_amdgcn_wmma_f32_16x16x32_bf16(false, a0, false, bfrag,
                                                       (short)0, acc0, false, false);
        acc1 = __builtin_amdgcn_wmma_f32_16x16x32_bf16(false, a1, false, bfrag,
                                                       (short)0, acc1, false, false);
    }

    // ---------------- stage 3: bias + scatter store ----------------
    const float bv = bias[hw * COUTC + o];
#pragma unroll
    for (int r = 0; r < 8; ++r) {
        const int m = r + 8 * half;         // C/D layout: VGPR r, lane half
        out[(size_t)((m      * COUTC + o)) * (HOUT * WOUT) + hw] = acc0[r] + bv;
        out[(size_t)(((16+m) * COUTC + o)) * (HOUT * WOUT) + hw] = acc1[r] + bv;
    }
}

extern "C" void kernel_launch(void* const* d_in, const int* in_sizes, int n_in,
                              void* d_out, int out_size, void* d_ws, size_t ws_size,
                              hipStream_t stream) {
    const float* feat = (const float*)d_in[0];  // [32,64,64,64]
    const float* wgt  = (const float*)d_in[1];  // [62,62,64,64,3,3]
    const float* bias = (const float*)d_in[2];  // [62,62,64]
    float* out = (float*)d_out;                 // [32,64,62,62]

    dim3 grid(WOUT, HOUT);   // one block per output pixel = one small GEMM
    dim3 block(128);         // 4 waves: one cout tile each, both batch tiles
    lc2d_wmma_kernel<<<grid, block, 0, stream>>>(feat, wgt, bias, out);
}
